// MultiHeadAttention_20229295964279
// MI455X (gfx1250) — compile-verified, matrix operand order checked by reading, not run on hardware
//
#include <hip/hip_runtime.h>
#include <hip/hip_bf16.h>

typedef __attribute__((ext_vector_type(16))) __bf16 v16bf;
typedef __attribute__((ext_vector_type(8)))  float  v8f;
typedef __attribute__((ext_vector_type(4)))  unsigned int v4u;
typedef __attribute__((ext_vector_type(8)))  int v8i;
typedef __attribute__((ext_vector_type(4)))  int v4i;

union Frag16 { unsigned int u[8]; v16bf v; };

#ifndef __has_builtin
#define __has_builtin(x) 0
#endif
#if __has_builtin(__builtin_amdgcn_tensor_load_to_lds) && __has_builtin(__builtin_amdgcn_s_wait_tensorcnt)
#define USE_TDM 1
#else
#define USE_TDM 0
#endif

__device__ __forceinline__ unsigned short f2bf(float f) {
    unsigned int u = __builtin_bit_cast(unsigned int, f);
    unsigned int r = (u + 0x7FFFu + ((u >> 16) & 1u)) >> 16;
    return (unsigned short)r;
}

__device__ __forceinline__ int kpair_idx(int i) {
    // 16-bit A/B fragment: VGPR i holds K pair {kidx, kidx+1}
    return 2 * i + (i >= 4 ? 8 : 0);
}

#if USE_TDM
// TDM: DMA a 32x64 bf16 tile (row-major, row stride 64 elems) from global to
// LDS, inserting 16B pad after every 128B row -> LDS row stride 144B (72 us).
// D# packing per CDNA5 ISA ch.10.7/8: group0 {count, lds_addr, gaddr, type=2},
// group1 {data_size=2B, pad_enable, pad_interval=32dw, pad_amount=4dw,
//         tensor 64x32, tile 64x32, dim0 stride 64}.
__device__ __forceinline__ void tdm_load_tile32x64(unsigned lds_byte_off,
                                                   const unsigned short* gsrc) {
    unsigned long long ga = (unsigned long long)(size_t)(const void*)gsrc;
    v4u g0;
    g0.x = 1u;                                               // count=1, no gather
    g0.y = lds_byte_off;                                     // lds_addr
    g0.z = (unsigned)(ga & 0xffffffffu);                     // global_addr[31:0]
    g0.w = (unsigned)((ga >> 32) & 0x01ffffffu) | (2u << 30);// gaddr[56:32] | type=2

    v8i g1;
    g1[0] = (int)((1u << 16)        // data_size = 1 -> 2 bytes
                | (1u << 20)        // pad_enable
                | (4u << 22)        // pad_interval: 4 -> every 32 DWORDs (128B)
                | (3u << 25));      // pad_amount:   3 -> 4 DWORDs (16B)
    g1[1] = (int)(64u << 16);       // tensor_dim0 = 64 (bits 79:48 low half)
    g1[2] = (int)(32u << 16);       // tensor_dim1 = 32 (bits 111:80 low half)
    g1[3] = (int)(64u << 16);       // tile_dim0 = 64   (bits 127:112)
    g1[4] = 32;                     // tile_dim1 = 32   (bits 143:128)
    g1[5] = 64;                     // tensor_dim0_stride = 64 (bits 207:160 low)
    g1[6] = 0;
    g1[7] = 0;

    v4i z4 = {0, 0, 0, 0};
#if defined(__clang_major__) && (__clang_major__ >= 23)
    v8i z8 = {0, 0, 0, 0, 0, 0, 0, 0};
    __builtin_amdgcn_tensor_load_to_lds(g0, g1, z4, z4, z8, 0);
#else
    __builtin_amdgcn_tensor_load_to_lds(g0, g1, z4, z4, 0);
#endif
}
#endif

// ---------------------------------------------------------------------------
// Generic GEMM: Y[M,Nn] = A[M,K] @ W[Nn,K]^T, epilogue (acc + bias[n]) * scale
// ---------------------------------------------------------------------------
#define BM 128
#define BN 128
#define BK 32
#define PADA 36   // ushorts per LDS tile row (32 data + 4 pad), 72B rows

template<bool A_BF16, bool HEADIFY>
__global__ __launch_bounds__(256)
void gemm_wmma(const void* __restrict__ Aptr, const float* __restrict__ W,
               const float* __restrict__ bias, void* __restrict__ outp,
               int M, int Nn, int K, float scale)
{
    __shared__ unsigned short sA[BM * PADA];
    __shared__ unsigned short sB[BN * PADA];

    const int t     = threadIdx.x;
    const int lane  = t & 31;
    const int wave  = t >> 5;
    const int waveM = wave & 3;
    const int waveN = wave >> 2;
    const int half  = lane >> 4;
    const int l16   = lane & 15;

    const int rowBase = blockIdx.y * BM;
    const int colBase = blockIdx.x * BN;

    v8f acc[2][4];
#pragma unroll
    for (int mg = 0; mg < 2; ++mg)
#pragma unroll
        for (int ng = 0; ng < 4; ++ng)
#pragma unroll
            for (int r = 0; r < 8; ++r) acc[mg][ng][r] = 0.0f;

    for (int k0 = 0; k0 < K; k0 += BK) {
        __syncthreads();
        if (A_BF16) {
            const unsigned short* A = (const unsigned short*)Aptr;
#pragma unroll
            for (int p = 0; p < 2; ++p) {
                int idx = p * 256 + t;
                int r = idx >> 2, seg = idx & 3;
                uint4 val = *(const uint4*)(A + (size_t)(rowBase + r) * K + k0 + seg * 8);
                unsigned short* d = &sA[r * PADA + seg * 8];
                *(uint2*)(d)     = make_uint2(val.x, val.y);
                *(uint2*)(d + 4) = make_uint2(val.z, val.w);
            }
        } else {
            const float* A = (const float*)Aptr;
#pragma unroll
            for (int p = 0; p < 4; ++p) {
                int idx = p * 256 + t;
                int r = idx >> 3, seg = idx & 7;
                float4 val = *(const float4*)(A + (size_t)(rowBase + r) * K + k0 + seg * 4);
                unsigned int lo = (unsigned int)f2bf(val.x) | ((unsigned int)f2bf(val.y) << 16);
                unsigned int hi = (unsigned int)f2bf(val.z) | ((unsigned int)f2bf(val.w) << 16);
                *(uint2*)&sA[r * PADA + seg * 4] = make_uint2(lo, hi);
            }
        }
#pragma unroll
        for (int p = 0; p < 4; ++p) {
            int idx = p * 256 + t;
            int r = idx >> 3, seg = idx & 7;
            float4 val = *(const float4*)(W + (size_t)(colBase + r) * K + k0 + seg * 4);
            unsigned int lo = (unsigned int)f2bf(val.x) | ((unsigned int)f2bf(val.y) << 16);
            unsigned int hi = (unsigned int)f2bf(val.z) | ((unsigned int)f2bf(val.w) << 16);
            *(uint2*)&sB[r * PADA + seg * 4] = make_uint2(lo, hi);
        }
        // prefetch next k-tile rows into cache (global_prefetch_b8)
        if (k0 + BK < K) {
            int pr = t & 127;
            if (A_BF16)
                __builtin_prefetch((const unsigned short*)Aptr + (size_t)(rowBase + pr) * K + k0 + BK, 0, 1);
            else
                __builtin_prefetch((const float*)Aptr + (size_t)(rowBase + pr) * K + k0 + BK, 0, 1);
            __builtin_prefetch(W + (size_t)(colBase + pr) * K + k0 + BK, 0, 1);
        }
        __syncthreads();

        Frag16 a[2];
#pragma unroll
        for (int mg = 0; mg < 2; ++mg) {
            const unsigned short* src = &sA[(waveM * 32 + mg * 16 + l16) * PADA + half * 8];
#pragma unroll
            for (int i = 0; i < 8; ++i)
                a[mg].u[i] = *(const unsigned int*)(src + kpair_idx(i));
        }
#pragma unroll
        for (int ng = 0; ng < 4; ++ng) {
            Frag16 b;
            const unsigned short* src = &sB[(waveN * 64 + ng * 16 + l16) * PADA + half * 8];
#pragma unroll
            for (int i = 0; i < 8; ++i)
                b.u[i] = *(const unsigned int*)(src + kpair_idx(i));
#pragma unroll
            for (int mg = 0; mg < 2; ++mg)
                acc[mg][ng] = __builtin_amdgcn_wmma_f32_16x16x32_bf16(
                    false, a[mg].v, false, b.v, (short)0, acc[mg][ng], false, false);
        }
    }

#pragma unroll
    for (int mg = 0; mg < 2; ++mg)
#pragma unroll
        for (int ng = 0; ng < 4; ++ng)
#pragma unroll
            for (int r = 0; r < 8; ++r) {
                int m = rowBase + waveM * 32 + mg * 16 + half * 8 + r;
                int n = colBase + waveN * 64 + ng * 16 + l16;
                float vout = (acc[mg][ng][r] + bias[n]) * scale;
                if (HEADIFY) {
                    int b = m >> 11, s = m & 2047;   // C = 2048
                    int h = n >> 6,  d = n & 63;     // head_dim = 64
                    ((unsigned short*)outp)[((size_t)(b * 16 + h) << 17) + ((size_t)s << 6) + d] = f2bf(vout);
                } else {
                    ((float*)outp)[(size_t)m * Nn + n] = vout;
                }
            }
}

// ---------------------------------------------------------------------------
// Flash attention, TDM double-buffered K/V tiles.
// Block = one (b,h) x 64 query rows; 4 waves x 16 rows each.
// ---------------------------------------------------------------------------
#define NT   32
#define KVP  72    // padded LDS stride (144B rows) — matches TDM pad descriptor
#define PP   34    // P-buffer stride
#define TILE_US (NT * KVP)            // ushorts per K or V tile
#define BUF_US  (2 * TILE_US)         // K+V per buffer

__global__ __launch_bounds__(128)
void attn_wmma(const unsigned short* __restrict__ Qh,
               const unsigned short* __restrict__ Kh,
               const unsigned short* __restrict__ Vh,
               unsigned short* __restrict__ Xout,
               int C, int N)
{
    // single shared block: [buf0: K|V][buf1: K|V][P per wave]
    __shared__ unsigned short sLds[2 * BUF_US + 4 * 16 * PP];

    const int t    = threadIdx.x;
    const int lane = t & 31;
    const int wave = t >> 5;
    const int half = lane >> 4;
    const int l16  = lane & 15;

    const int nCblk = C >> 6;
    const int bh    = blockIdx.x / nCblk;
    const int cblk  = blockIdx.x % nCblk;
    const int b     = bh >> 4;
    const int h     = bh & 15;

    const unsigned short* Qbh = Qh + (size_t)bh * C * 64;
    const unsigned short* Kbh = Kh + (size_t)bh * N * 64;
    const unsigned short* Vbh = Vh + (size_t)bh * N * 64;

    const int qRow0 = cblk * 64 + wave * 16;

    // Q A-fragments (2 k-steps over d=64)
    Frag16 qf[2];
    {
        const unsigned short* qsrc = Qbh + (size_t)(qRow0 + l16) * 64 + half * 8;
#pragma unroll
        for (int ks = 0; ks < 2; ++ks)
#pragma unroll
            for (int i = 0; i < 8; ++i)
                qf[ks].u[i] = *(const unsigned int*)(qsrc + ks * 32 + kpair_idx(i));
    }

    v8f xacc[4];
#pragma unroll
    for (int cg = 0; cg < 4; ++cg)
#pragma unroll
        for (int r = 0; r < 8; ++r) xacc[cg][r] = 0.0f;
    float mrun[8], lrun[8];
#pragma unroll
    for (int r = 0; r < 8; ++r) { mrun[r] = -1e30f; lrun[r] = 0.0f; }

    unsigned short* sP = sLds + 2 * BUF_US + wave * 16 * PP;
    const int NTILES = N / NT;

#if USE_TDM
    const unsigned ldsBase = (unsigned)(size_t)(void*)sLds; // LDS byte offset
    if (wave == 0) {
        tdm_load_tile32x64(ldsBase,                Kbh);
        tdm_load_tile32x64(ldsBase + TILE_US * 2,  Vbh);
    }
#endif

    for (int nt = 0; nt < NTILES; ++nt) {
        const int buf = nt & 1;
        unsigned short* sK = sLds + buf * BUF_US;
        unsigned short* sV = sK + TILE_US;

        __syncthreads();   // all waves done reading both buffers' prior contents
#if USE_TDM
        if (wave == 0) {
            if (nt + 1 < NTILES) {
                const int nbuf = buf ^ 1;
                const size_t goff = (size_t)(nt + 1) * NT * 64;
                tdm_load_tile32x64(ldsBase + (unsigned)(nbuf * BUF_US) * 2u,           Kbh + goff);
                tdm_load_tile32x64(ldsBase + (unsigned)(nbuf * BUF_US + TILE_US) * 2u, Vbh + goff);
                __builtin_amdgcn_s_wait_tensorcnt(2);  // current tile's 2 DMAs done
            } else {
                __builtin_amdgcn_s_wait_tensorcnt(0);
            }
        }
#else
#pragma unroll
        for (int p = 0; p < 2; ++p) {
            int idx = p * 128 + t;
            int r = idx >> 3, seg = idx & 7;
            *(uint4*)&sK[r * KVP + seg * 8] = *(const uint4*)(Kbh + (size_t)(nt * NT + r) * 64 + seg * 8);
            *(uint4*)&sV[r * KVP + seg * 8] = *(const uint4*)(Vbh + (size_t)(nt * NT + r) * 64 + seg * 8);
        }
#endif
        __syncthreads();   // tile data visible to all waves

        // S[16x32] = Q[16x64] @ K_tile[32x64]^T
        v8f s0, s1;
#pragma unroll
        for (int r = 0; r < 8; ++r) { s0[r] = 0.0f; s1[r] = 0.0f; }
#pragma unroll
        for (int ks = 0; ks < 2; ++ks) {
            Frag16 b0, b1;
            const unsigned short* p0 = &sK[(0  + l16) * KVP + half * 8 + ks * 32];
            const unsigned short* p1 = &sK[(16 + l16) * KVP + half * 8 + ks * 32];
#pragma unroll
            for (int i = 0; i < 8; ++i) {
                int kidx = kpair_idx(i);
                b0.u[i] = *(const unsigned int*)(p0 + kidx);
                b1.u[i] = *(const unsigned int*)(p1 + kidx);
            }
            s0 = __builtin_amdgcn_wmma_f32_16x16x32_bf16(false, qf[ks].v, false, b0.v, (short)0, s0, false, false);
            s1 = __builtin_amdgcn_wmma_f32_16x16x32_bf16(false, qf[ks].v, false, b1.v, (short)0, s1, false, false);
        }

        // online softmax (row r+8*half; butterfly masks <=8 stay in 16-lane half)
#pragma unroll
        for (int r = 0; r < 8; ++r) {
            float mx = fmaxf(s0[r], s1[r]);
#pragma unroll
            for (int msk = 8; msk >= 1; msk >>= 1)
                mx = fmaxf(mx, __shfl_xor(mx, msk, 32));
            float nm = fmaxf(mrun[r], mx);
            float corr = __expf(mrun[r] - nm);
            mrun[r] = nm;
            float e0 = __expf(s0[r] - nm);
            float e1 = __expf(s1[r] - nm);
            float ps = e0 + e1;
#pragma unroll
            for (int msk = 8; msk >= 1; msk >>= 1)
                ps += __shfl_xor(ps, msk, 32);
            lrun[r] = lrun[r] * corr + ps;
            int mrow = r + 8 * half;
            sP[mrow * PP + l16]      = f2bf(e0);
            sP[mrow * PP + 16 + l16] = f2bf(e1);
#pragma unroll
            for (int cg = 0; cg < 4; ++cg) xacc[cg][r] *= corr;
        }
        asm volatile("s_wait_dscnt 0" ::: "memory");   // per-wave P store->load

        // P as A-fragment (16x32 over k=n_local)
        Frag16 pf;
        {
            const unsigned short* psrc = &sP[l16 * PP + half * 8];
#pragma unroll
            for (int i = 0; i < 8; ++i)
                pf.u[i] = *(const unsigned int*)(psrc + kpair_idx(i));
        }
        // x[16x64] += P[16x32] @ V_tile[32x64]
#pragma unroll
        for (int cg = 0; cg < 4; ++cg) {
            Frag16 bv;
#pragma unroll
            for (int i = 0; i < 8; ++i) {
                int kk = kpair_idx(i) + 8 * half;
                int d  = cg * 16 + l16;
                unsigned int lo = sV[kk * KVP + d];
                unsigned int hi = sV[(kk + 1) * KVP + d];
                bv.u[i] = lo | (hi << 16);
            }
            xacc[cg] = __builtin_amdgcn_wmma_f32_16x16x32_bf16(
                false, pf.v, false, bv.v, (short)0, xacc[cg], false, false);
        }
    }

    // finalize: divide by l, write bf16 to merged [B*C, 1024] layout
#pragma unroll
    for (int cg = 0; cg < 4; ++cg)
#pragma unroll
        for (int r = 0; r < 8; ++r) {
            float vv = xacc[cg][r] / lrun[r];
            int c = qRow0 + r + 8 * half;
            int d = cg * 16 + l16;
            Xout[((size_t)b * C + c) * 1024 + h * 64 + d] = f2bf(vv);
        }
}

// ---------------------------------------------------------------------------
extern "C" void kernel_launch(void* const* d_in, const int* in_sizes, int n_in,
                              void* d_out, int out_size, void* d_ws, size_t ws_size,
                              hipStream_t stream) {
    (void)in_sizes; (void)n_in; (void)out_size; (void)ws_size;
    const float* q  = (const float*)d_in[0];
    const float* k  = (const float*)d_in[1];
    const float* v  = (const float*)d_in[2];
    const float* Wq = (const float*)d_in[3];
    const float* bq = (const float*)d_in[4];
    const float* Wk = (const float*)d_in[5];
    const float* bk = (const float*)d_in[6];
    const float* Wv = (const float*)d_in[7];
    const float* bv = (const float*)d_in[8];
    const float* Wo = (const float*)d_in[9];
    const float* bo = (const float*)d_in[10];
    float* out = (float*)d_out;

    const int B = 2, C = 2048, N = 2048, K = 1024, H = 1024;
    const int M = B * C;   // 4096

    unsigned short* Qh = (unsigned short*)d_ws;          // [32][2048][64] bf16
    unsigned short* Kh = Qh + (size_t)M * H;
    unsigned short* Vh = Kh + (size_t)M * H;
    unsigned short* Xa = Vh + (size_t)M * H;             // [4096][1024] bf16

    dim3 gb(H / BN, M / BM);
    const float scale = 0.125f;  // 1/sqrt(64)
    gemm_wmma<false, true ><<<gb, 256, 0, stream>>>(q, Wq, bq, Qh, M, H, K, scale);
    gemm_wmma<false, true ><<<gb, 256, 0, stream>>>(k, Wk, bk, Kh, M, H, K, 1.0f);
    gemm_wmma<false, true ><<<gb, 256, 0, stream>>>(v, Wv, bv, Vh, M, H, K, 1.0f);
    attn_wmma<<<B * 16 * (C / 64), 128, 0, stream>>>(Qh, Kh, Vh, Xa, C, N);
    gemm_wmma<true,  false><<<gb, 256, 0, stream>>>(Xa, Wo, bo, out, M, H, K, 1.0f);
}
